// VQVAE3D_27857157881937
// MI455X (gfx1250) — compile-verified, matrix operand order checked by reading, not run on hardware
//
#include <hip/hip_runtime.h>
#include <hip/hip_bf16.h>
#include <cstdint>
#include <cstddef>

// ---------------------------------------------------------------------------
// VQ-VAE 3D forward for gfx1250 (MI455X).
// Conv / deconv / VQ distances as implicit-GEMM via v_wmma_f32_16x16x32_bf16.
// A-tile: branchless implicit-im2col gather (row decode hoisted out of K loop).
// B-tile: tile-major packed bf16 weights copied with global_load_async_to_lds.
// ---------------------------------------------------------------------------

typedef __bf16 bf16_t;
typedef __bf16 v8bf  __attribute__((ext_vector_type(8)));
typedef __bf16 v16bf __attribute__((ext_vector_type(16)));
typedef float  v8f   __attribute__((ext_vector_type(8)));

#define BM 128
#define BN 32
#define BK 32

struct GemmArgs {
  const float*  __restrict__ A;    // source activation tensor (fp32)
  const bf16_t* __restrict__ Bp;   // packed weights [class][kblk][N][32]
  const float*  __restrict__ bias; // per-output-channel bias (unused for VQ)
  float*        __restrict__ out;  // destination tensor
  int M, N, K;          // M = rows (per parity class for deconv); M % 128 == 0
  int IC, ID, IH, IW;   // input channel / spatial dims
  int OC;               // output channels
  int lOD, lOH, lOW;    // log2 of output spatial dims
};

// MODE: 0 = conv k4 s2 p1, 1 = deconv k4 s2 p1 (per output-parity class),
//       2 = VQ dot (A = latent [B][K][S] transpose-read, C row-major)
// ACT : 0 none, 1 relu, 2 sigmoid
template <int MODE, int ACT>
__global__ void __launch_bounds__(256)
gemm_wmma(GemmArgs g) {
  __shared__ __align__(16) bf16_t As[BM][BK + 8];
  __shared__ __align__(16) bf16_t Bs[BN][BK + 8];

  const int tid  = threadIdx.x;
  const int lane = tid & 31;
  const int wave = tid >> 5;
  const int wm   = wave >> 1;   // 0..3  (M sub-tile)
  const int wn   = wave & 1;    // 0..1  (N sub-tile)
  const int hf   = lane >> 4;
  const int l15  = lane & 15;

  const int row0 = blockIdx.x * BM;
  const int col0 = blockIdx.y * BN;
  const int cls  = blockIdx.z;

  int pz = 0, py = 0, px = 0;
  if (MODE == 1) { pz = (cls >> 2) & 1; py = (cls >> 1) & 1; px = cls & 1; }

  const bf16_t* __restrict__ Bsrc =
      g.Bp + (MODE == 1 ? (size_t)cls * (size_t)g.K * (size_t)g.N : 0);

  // ---- A staging ownership: one tile row + 16 consecutive k per thread ----
  const int am   = tid >> 1;
  const int ak   = (tid & 1) << 4;
  const int arow = row0 + am;

  // K-invariant per-thread decode
  const int lS = g.lOD + g.lOH + g.lOW;
  int an = 0, aid0 = 0, aih0 = 0, aiw0 = 0;  // conv
  size_t vqbase = 0;                         // vq
  int tapoff[8]; unsigned tapok = 0;         // deconv
  const size_t chanvol = (size_t)g.ID * g.IH * g.IW;
  if (MODE == 0) {
    an   = arow >> lS;
    aid0 = (((arow >> (g.lOH + g.lOW)) & ((1 << g.lOD) - 1)) << 1) - 1;
    aih0 = (((arow >> g.lOW) & ((1 << g.lOH) - 1)) << 1) - 1;
    aiw0 = ((arow & ((1 << g.lOW) - 1)) << 1) - 1;
  } else if (MODE == 1) {
    const int l2D = g.lOD - 1, l2H = g.lOH - 1, l2W = g.lOW - 1;
    an = arow >> (l2D + l2H + l2W);
    const int zc = (arow >> (l2H + l2W)) & ((1 << l2D) - 1);
    const int yc = (arow >> l2W) & ((1 << l2H) - 1);
    const int xc = arow & ((1 << l2W) - 1);
    const int z = 2 * zc + pz, y = 2 * yc + py, xx = 2 * xc + px;
#pragma unroll
    for (int t = 0; t < 8; ++t) {
      const int kd = (1 - pz) + (((t >> 2) & 1) << 1);
      const int kh = (1 - py) + (((t >> 1) & 1) << 1);
      const int kw = (1 - px) + ((t & 1) << 1);
      const int dz = z + 1 - kd, dy = y + 1 - kh, dx = xx + 1 - kw;
      const int id = dz >> 1, ih = dy >> 1, iw = dx >> 1;
      const bool ok = dz >= 0 && id < g.ID && dy >= 0 && ih < g.IH &&
                      dx >= 0 && iw < g.IW;
      tapoff[t] = ok ? (id * g.IH + ih) * g.IW + iw : 0;
      tapok |= (unsigned)ok << t;
    }
  } else {
    const int n = arow >> lS;
    const int s = arow & ((1 << lS) - 1);
    vqbase = ((size_t)n * g.K << lS) + s;
  }

  // ---- B staging ownership: one 8-byte chunk per thread (async copy) ----
  const int bn = tid >> 3;            // 0..31 : LDS row (column of B)
  const int bc = (tid & 7) << 2;      // element offset within row
  int bcol = col0 + bn;
  if (bcol >= g.N) bcol = g.N - 1;    // clamp: stays in-bounds, never stored
  const unsigned blds = (unsigned)(uintptr_t)&Bs[bn][bc];

  v8f acc[2] = {};

  int kk = 0;
  do {
    // ---- B tile: raw async copy, ASYNCcnt-tracked ----
    {
      const bf16_t* gp = Bsrc + (((size_t)(kk >> 5) * g.N + bcol) << 5) + bc;
      asm volatile("global_load_async_to_lds_b64 %0, %1, off"
                   :: "v"(blds), "v"(gp) : "memory");
    }
    // ---- A tile: branchless implicit-im2col gather ----
    bf16_t vals[16];
    if (MODE == 0) {
      const int kg0 = kk + ak;          // 16-aligned => (ic, kd) fixed
      const int ic  = kg0 >> 6;
      const int kd  = (kg0 >> 4) & 3;
      const int id  = aid0 + kd;
      const bool idok = (unsigned)id < (unsigned)g.ID;
      const float* __restrict__ base =
          g.A + ((size_t)(an * g.IC + ic) * g.ID + (idok ? id : 0)) *
                    ((size_t)g.IH * g.IW);
#pragma unroll
      for (int j = 0; j < 16; ++j) {
        const int ih = aih0 + (j >> 2);
        const int iw = aiw0 + (j & 3);
        const bool ok = idok && (unsigned)ih < (unsigned)g.IH &&
                        (unsigned)iw < (unsigned)g.IW;
        const float v = base[ok ? ih * g.IW + iw : 0];
        vals[j] = (bf16_t)(ok ? v : 0.f);
      }
    } else if (MODE == 1) {
      const int ic0 = (kk + ak) >> 3;   // 16 elems span ic0, ic0+1
#pragma unroll
      for (int j = 0; j < 16; ++j) {
        const int t = j & 7;
        const float* __restrict__ base =
            g.A + ((size_t)an * g.IC + ic0 + (j >> 3)) * chanvol;
        const float v = base[tapoff[t]];
        vals[j] = (bf16_t)(((tapok >> t) & 1) ? v : 0.f);
      }
    } else {
#pragma unroll
      for (int j = 0; j < 16; ++j)
        vals[j] = (bf16_t)g.A[vqbase + ((size_t)(kk + ak + j) << lS)];
    }
    v8bf p0, p1;
#pragma unroll
    for (int j = 0; j < 8; ++j) { p0[j] = vals[j]; p1[j] = vals[j + 8]; }
    *(v8bf*)&As[am][ak]     = p0;      // two ds_store_b128
    *(v8bf*)&As[am][ak + 8] = p1;

    asm volatile("s_wait_asynccnt 0x0" ::: "memory");
    __syncthreads();

    // ---- fragments from LDS (ds_load_b128 pairs) + WMMA ----
    {
      const int k0 = hf * 8;
      const int k1 = 16 + hf * 8;

      v8bf blo = *(const v8bf*)&Bs[wn * 16 + l15][k0];
      v8bf bhi = *(const v8bf*)&Bs[wn * 16 + l15][k1];
      v16bf bf = __builtin_shufflevector(blo, bhi, 0, 1, 2, 3, 4, 5, 6, 7, 8, 9,
                                         10, 11, 12, 13, 14, 15);

      v8bf a0l = *(const v8bf*)&As[wm * 32 + l15][k0];
      v8bf a0h = *(const v8bf*)&As[wm * 32 + l15][k1];
      v16bf a0 = __builtin_shufflevector(a0l, a0h, 0, 1, 2, 3, 4, 5, 6, 7, 8, 9,
                                         10, 11, 12, 13, 14, 15);
      v8bf a1l = *(const v8bf*)&As[wm * 32 + 16 + l15][k0];
      v8bf a1h = *(const v8bf*)&As[wm * 32 + 16 + l15][k1];
      v16bf a1 = __builtin_shufflevector(a1l, a1h, 0, 1, 2, 3, 4, 5, 6, 7, 8, 9,
                                         10, 11, 12, 13, 14, 15);

      acc[0] = __builtin_amdgcn_wmma_f32_16x16x32_bf16(
          false, a0, false, bf, (short)0, acc[0], false, false);
      acc[1] = __builtin_amdgcn_wmma_f32_16x16x32_bf16(
          false, a1, false, bf, (short)0, acc[1], false, false);
    }
    __syncthreads();
    kk += BK;
  } while (kk < g.K);

  // ---- epilogue: bias + activation, direct NCDHW (or row-major) store ----
  const int col = col0 + wn * 16 + l15;
  if (col < g.N) {
    const float bv = (MODE == 2) ? 0.f : g.bias[col];
    // MODE 0/2: the 32-row sub-tile never crosses an n boundary -> hoist base
    size_t addr0 = 0;
    if (MODE == 0) {
      const int srow = row0 + wm * 32;
      const int n = srow >> lS;
      addr0 = (((size_t)(n * g.OC + col)) << lS) + (srow & ((1 << lS) - 1));
    } else if (MODE == 2) {
      addr0 = (size_t)(row0 + wm * 32) * g.N + col;
    }
#pragma unroll
    for (int f = 0; f < 2; ++f) {
#pragma unroll
      for (int j = 0; j < 8; ++j) {
        const int delta = f * 16 + hf * 8 + j;
        float v = acc[f][j] + bv;
        if (ACT == 1) v = fmaxf(v, 0.f);
        if (ACT == 2) v = __builtin_amdgcn_rcpf(1.f + __expf(-v));
        size_t addr;
        if (MODE == 2) {
          addr = addr0 + (size_t)delta * g.N;
        } else if (MODE == 0) {
          addr = addr0 + delta;
        } else {
          const int row = row0 + wm * 32 + delta;
          const int l2D = g.lOD - 1, l2H = g.lOH - 1, l2W = g.lOW - 1;
          const int n  = row >> (l2D + l2H + l2W);
          const int zc = (row >> (l2H + l2W)) & ((1 << l2D) - 1);
          const int yc = (row >> l2W) & ((1 << l2H) - 1);
          const int xc = row & ((1 << l2W) - 1);
          const int z = 2 * zc + pz, y = 2 * yc + py, xx = 2 * xc + px;
          addr = (((size_t)(n * g.OC + col)) << lS) +
                 ((size_t)((z << g.lOH) + y) << g.lOW) + xx;
        }
        g.out[addr] = v;
      }
    }
  }
}

// ---------------- weight / codebook packing (fp32 -> bf16) ------------------
// All packers emit the LDS tile-major layout: [kblk][n][k&31].

// conv weight (OC,IC,4,4,4) -> tile-major [K=IC*64][OC]
__global__ void pack_conv_w(const float* __restrict__ w,
                            bf16_t* __restrict__ Bp, int IC, int OC) {
  const int total = IC * 64 * OC;
  const int i = blockIdx.x * blockDim.x + threadIdx.x;
  if (i >= total) return;
  const int k = i / OC, oc = i % OC;
  const int ic = k >> 6, t = k & 63;
  const size_t j = (((size_t)(k >> 5) * OC + oc) << 5) + (k & 31);
  Bp[j] = (bf16_t)w[((size_t)oc * IC + ic) * 64 + t];
}

// deconv weight (IC,OC,4,4,4) -> per parity class: tile-major [K=IC*8][OC]
__global__ void pack_deconv_w(const float* __restrict__ w,
                              bf16_t* __restrict__ Bp, int IC, int OC) {
  const int KN = IC * 8 * OC;
  const int total = 8 * KN;
  const int i = blockIdx.x * blockDim.x + threadIdx.x;
  if (i >= total) return;
  const int p = i / KN, r = i % KN;
  const int k = r / OC, oc = r % OC;
  const int pz = (p >> 2) & 1, py = (p >> 1) & 1, px = p & 1;
  const int ic = k >> 3;
  const int kd = (1 - pz) + (((k >> 2) & 1) << 1);
  const int kh = (1 - py) + (((k >> 1) & 1) << 1);
  const int kw = (1 - px) + ((k & 1) << 1);
  const size_t j =
      (size_t)p * KN + (((size_t)(k >> 5) * OC + oc) << 5) + (k & 31);
  Bp[j] = (bf16_t)w[((((size_t)ic * OC + oc) * 4 + kd) * 4 + kh) * 4 + kw];
}

// codebook (Kc,D) -> tile-major [K=D][Kc] (transpose)
__global__ void pack_codebook(const float* __restrict__ cb,
                              bf16_t* __restrict__ Bp, int Kc, int D) {
  const int total = Kc * D;
  const int i = blockIdx.x * blockDim.x + threadIdx.x;
  if (i >= total) return;
  const int d = i / Kc, c = i % Kc;
  const size_t j = (((size_t)(d >> 5) * Kc + c) << 5) + (d & 31);
  Bp[j] = (bf16_t)cb[(size_t)c * D + d];
}

__global__ void vq_cnorm(const float* __restrict__ cb, float* __restrict__ cn,
                         int Kc, int D) {
  const int c = blockIdx.x * blockDim.x + threadIdx.x;
  if (c >= Kc) return;
  const float* p = cb + (size_t)c * D;
  float s = 0.f;
  for (int k = 0; k < D; ++k) s += p[k] * p[k];
  cn[c] = s;
}

// wave-per-row argmin over codes of (||c||^2 - 2 * z.c)
__global__ void vq_argmin(const float* __restrict__ P,
                          const float* __restrict__ cn, int* __restrict__ idx,
                          int M, int Kc) {
  const int r    = (blockIdx.x * blockDim.x + threadIdx.x) >> 5;
  const int lane = threadIdx.x & 31;
  if (r >= M) return;
  const float* p = P + (size_t)r * Kc;
  float best = 3.4e38f;
  int bi = 0;
  for (int c = lane; c < Kc; c += 32) {
    const float s = cn[c] - 2.f * p[c];
    if (s < best) { best = s; bi = c; }
  }
#pragma unroll
  for (int d = 16; d; d >>= 1) {
    const float ob = __shfl_xor(best, d, 32);
    const int   oi = __shfl_xor(bi, d, 32);
    if (ob < best || (ob == best && oi < bi)) { best = ob; bi = oi; }
  }
  if (lane == 0) idx[r] = bi;
}

// quantized[b][d][s] = codebook[idx[b*S+s]][d]
__global__ void vq_gather(const int* __restrict__ idx,
                          const float* __restrict__ cb, float* __restrict__ q,
                          int lD, int lS, int total) {
  const int i = blockIdx.x * blockDim.x + threadIdx.x;
  if (i >= total) return;
  const int b = i >> (lD + lS);
  const int d = (i >> lS) & ((1 << lD) - 1);
  const int s = i & ((1 << lS) - 1);
  const int code = idx[(b << lS) + s];
  q[i] = cb[((size_t)code << lD) + d];
}

// ---------------------------------------------------------------------------

static inline unsigned cdiv(unsigned a, unsigned b) { return (a + b - 1) / b; }

extern "C" void kernel_launch(void* const* d_in, const int* in_sizes, int n_in,
                              void* d_out, int out_size, void* d_ws,
                              size_t ws_size, hipStream_t stream) {
  (void)in_sizes; (void)n_in; (void)out_size; (void)ws_size;
  const float* x   = (const float*)d_in[0];
  const float* e1w = (const float*)d_in[1];
  const float* e1b = (const float*)d_in[2];
  const float* e2w = (const float*)d_in[3];
  const float* e2b = (const float*)d_in[4];
  const float* e3w = (const float*)d_in[5];
  const float* e3b = (const float*)d_in[6];
  const float* cb  = (const float*)d_in[7];
  const float* d1w = (const float*)d_in[8];
  const float* d1b = (const float*)d_in[9];
  const float* d2w = (const float*)d_in[10];
  const float* d2b = (const float*)d_in[11];
  const float* d3w = (const float*)d_in[12];
  const float* d3b = (const float*)d_in[13];

  // ---- output regions (x_hat | quantized | idx) ----
  float* out_xhat = (float*)d_out;                           // 4*1*128^3
  float* out_q    = out_xhat + (size_t)4 * 128 * 128 * 128;  // 4*512*16^3
  int*   out_idx  = (int*)(out_q + (size_t)4 * 512 * 4096);  // 4*4096

  // ---- workspace carve-out ----
  char* ws = (char*)d_ws;
  size_t off = 0;
  auto take = [&](size_t bytes) -> char* {
    char* p = ws + off;
    off = (off + bytes + 255) & ~(size_t)255;
    return p;
  };
  float*  h1  = (float*)take((size_t)4 * 16 * 262144 * 4);   // 64 MB
  float*  h2  = (float*)take((size_t)4 * 32 * 32768 * 4);    // 16 MB
  float*  z3  = (float*)take((size_t)4 * 512 * 4096 * 4);    // 32 MB
  float*  g1  = (float*)take((size_t)4 * 32 * 32768 * 4);    // 16 MB
  float*  g2  = (float*)take((size_t)4 * 16 * 262144 * 4);   // 64 MB
  float*  P   = (float*)take((size_t)16384 * 512 * 4);       // 32 MB
  float*  cn  = (float*)take(512 * 4);
  bf16_t* wb1 = (bf16_t*)take((size_t)64 * 16 * 2);
  bf16_t* wb2 = (bf16_t*)take((size_t)1024 * 32 * 2);
  bf16_t* wb3 = (bf16_t*)take((size_t)2048 * 512 * 2);
  bf16_t* cbp = (bf16_t*)take((size_t)512 * 512 * 2);
  bf16_t* wd1 = (bf16_t*)take((size_t)8 * 4096 * 32 * 2);
  bf16_t* wd2 = (bf16_t*)take((size_t)8 * 256 * 16 * 2);
  bf16_t* wd3 = (bf16_t*)take((size_t)8 * 128 * 1 * 2);

  // ---- pack weights / codebook ----
  pack_conv_w<<<cdiv(64 * 16, 256), 256, 0, stream>>>(e1w, wb1, 1, 16);
  pack_conv_w<<<cdiv(1024 * 32, 256), 256, 0, stream>>>(e2w, wb2, 16, 32);
  pack_conv_w<<<cdiv(2048 * 512, 256), 256, 0, stream>>>(e3w, wb3, 32, 512);
  pack_deconv_w<<<cdiv(8 * 4096 * 32, 256), 256, 0, stream>>>(d1w, wd1, 512, 32);
  pack_deconv_w<<<cdiv(8 * 256 * 16, 256), 256, 0, stream>>>(d2w, wd2, 32, 16);
  pack_deconv_w<<<cdiv(8 * 128 * 1, 256), 256, 0, stream>>>(d3w, wd3, 16, 1);
  pack_codebook<<<cdiv(512 * 512, 256), 256, 0, stream>>>(cb, cbp, 512, 512);
  vq_cnorm<<<cdiv(512, 256), 256, 0, stream>>>(cb, cn, 512, 512);

  // ---- encoder ----
  {
    GemmArgs a{x, wb1, e1b, h1, 4 * 262144, 16, 64, 1, 128, 128, 128, 16, 6, 6, 6};
    dim3 gr(cdiv(a.M, BM), cdiv(a.N, BN), 1);
    gemm_wmma<0, 1><<<gr, 256, 0, stream>>>(a);
  }
  {
    GemmArgs a{h1, wb2, e2b, h2, 4 * 32768, 32, 1024, 16, 64, 64, 64, 32, 5, 5, 5};
    dim3 gr(cdiv(a.M, BM), cdiv(a.N, BN), 1);
    gemm_wmma<0, 1><<<gr, 256, 0, stream>>>(a);
  }
  {
    GemmArgs a{h2, wb3, e3b, z3, 16384, 512, 2048, 32, 32, 32, 32, 512, 4, 4, 4};
    dim3 gr(cdiv(a.M, BM), cdiv(a.N, BN), 1);
    gemm_wmma<0, 0><<<gr, 256, 0, stream>>>(a);
  }

  // ---- vector quantization ----
  {
    GemmArgs a{z3, cbp, nullptr, P, 16384, 512, 512, 0, 0, 0, 0, 512, 4, 4, 4};
    dim3 gr(cdiv(a.M, BM), cdiv(a.N, BN), 1);
    gemm_wmma<2, 0><<<gr, 256, 0, stream>>>(a);
  }
  vq_argmin<<<cdiv(16384 * 32, 256), 256, 0, stream>>>(P, cn, out_idx, 16384,
                                                       512);
  vq_gather<<<cdiv(4 * 512 * 4096, 256), 256, 0, stream>>>(
      out_idx, cb, out_q, 9, 12, 4 * 512 * 4096);

  // ---- decoder (deconvs as 8 parity-class GEMMs each) ----
  {
    GemmArgs a{out_q, wd1, d1b, g1, 16384, 32, 4096, 512, 16, 16, 16, 32, 5, 5, 5};
    dim3 gr(cdiv(a.M, BM), cdiv(a.N, BN), 8);
    gemm_wmma<1, 1><<<gr, 256, 0, stream>>>(a);
  }
  {
    GemmArgs a{g1, wd2, d2b, g2, 131072, 16, 256, 32, 32, 32, 32, 16, 6, 6, 6};
    dim3 gr(cdiv(a.M, BM), cdiv(a.N, BN), 8);
    gemm_wmma<1, 1><<<gr, 256, 0, stream>>>(a);
  }
  {
    GemmArgs a{g2, wd3, d3b, out_xhat, 1048576, 1, 128, 16, 64, 64, 64, 1, 7, 7, 7};
    dim3 gr(cdiv(a.M, BM), cdiv(a.N, BN), 8);
    gemm_wmma<1, 2><<<gr, 256, 0, stream>>>(a);
  }
}
